// GATLinkEncoder_70076686401546
// MI455X (gfx1250) — compile-verified, matrix operand order checked by reading, not run on hardware
//
#include <hip/hip_runtime.h>
#include <hip/hip_bf16.h>

// ---------------------------------------------------------------------------
// GATLinkEncoder for MI455X (gfx1250, wave32, WMMA)
//   N=50000 nodes, E=400000 edges (+N self loops), EMB=128, HID=128, H=4, OUT=256
// GEMMs run on v_wmma_f32_16x16x32_bf16 (f32 accumulate); edge softmax/scatter
// uses f32 global atomics (RMW in the 192MB L2 - out1/H2 are L2-resident).
// ---------------------------------------------------------------------------

typedef __attribute__((ext_vector_type(16))) __bf16 v16bf;
typedef __attribute__((ext_vector_type(8)))  __bf16 v8bf;
typedef __attribute__((ext_vector_type(8)))  float  v8f;

__device__ __forceinline__ __bf16 f2bf(float x) {
  unsigned u = __float_as_uint(x);
  unsigned r = (u + 0x7FFFu + ((u >> 16) & 1u)) >> 16;   // round-to-nearest-even
  unsigned short h = (unsigned short)r;
  return __builtin_bit_cast(__bf16, h);
}

// ---------------- conversion / staging kernels ----------------

__global__ void k_f32_to_bf16(const float* __restrict__ in,
                              __bf16* __restrict__ out, long long n) {
  long long t = (long long)blockIdx.x * blockDim.x + threadIdx.x;
  if (t < n) out[t] = f2bf(in[t]);
}

// in: [R, C] row-major f32  ->  out: [C, R] row-major bf16 (i.e. transposed)
__global__ void k_transpose_bf16(const float* __restrict__ in,
                                 __bf16* __restrict__ out, int R, int C) {
  int t = blockIdx.x * blockDim.x + threadIdx.x;
  if (t >= R * C) return;
  int r = t / C, c = t - r * C;
  out[(size_t)c * R + r] = f2bf(in[t]);
}

__global__ void k_fill_u32(unsigned* __restrict__ p, unsigned v, long long n) {
  long long t = (long long)blockIdx.x * blockDim.x + threadIdx.x;
  if (t < n) p[t] = v;
}

// ---------------- WMMA GEMM:  C[M,Nc] = A[M,K] * B[K,Nc] ----------------
// A: [M,K] bf16 row-major.  Bt: [Nc,K] bf16 row-major (B transposed).
// One wave per 16x16 tile, K-steps of 32 via v_wmma_f32_16x16x32_bf16.
// Per-lane operand layout follows CDNA5 ISA 16-bit A/B tables:
//   lanes 0-15 : K {k..k+7, k+16..k+23},  lanes 16-31 : K {k+8..k+15, k+24..k+31}

__global__ void __launch_bounds__(128)
k_gemm_bf16_wmma(const __bf16* __restrict__ A, const __bf16* __restrict__ Bt,
                 float* __restrict__ C, int M, int K, int Nc) {
  const int wave  = threadIdx.x >> 5;
  const int lane  = threadIdx.x & 31;
  const int tileN = (blockIdx.x * 4 + wave) * 16;
  const int tileM = blockIdx.y * 16;
  if (tileN >= Nc) return;                       // wave-uniform exit
  const int r    = lane & 15;
  const int koff = (lane >> 4) << 3;             // 0 or 8
  int arow = tileM + r; if (arow >= M) arow = M - 1;   // clamped load row
  const __bf16* Ap = A  + (size_t)arow * K;
  const __bf16* Bp = Bt + (size_t)(tileN + r) * K;
  v8f acc = {};
  for (int k = 0; k < K; k += 32) {
    union { v16bf v; v8bf h[2]; } a, b;
    a.h[0] = *(const v8bf*)(Ap + k + koff);
    a.h[1] = *(const v8bf*)(Ap + k + 16 + koff);
    b.h[0] = *(const v8bf*)(Bp + k + koff);
    b.h[1] = *(const v8bf*)(Bp + k + 16 + koff);
    acc = __builtin_amdgcn_wmma_f32_16x16x32_bf16(false, a.v, false, b.v,
                                                  (short)0, acc, false, false);
  }
  const int mbase = tileM + ((lane >> 4) << 3);  // C layout: VGPR i -> row mbase+i
  const int col   = tileN + r;
#pragma unroll
  for (int i = 0; i < 8; ++i) {
    int row = mbase + i;
    if (row < M) C[(size_t)row * Nc + col] = acc[i];
  }
}

// ---------------- attention dot products ----------------
// out[n,h] = dot(X[n, h*C : (h+1)*C], a[h, :])   (one wave per (n,h))

__global__ void k_dot_att(const float* __restrict__ X, const float* __restrict__ a,
                          float* __restrict__ out, int N, int H, int C) {
  int warp = (int)((blockIdx.x * (long long)blockDim.x + threadIdx.x) >> 5);
  int lane = threadIdx.x & 31;
  if (warp >= N * H) return;
  int n = warp / H, h = warp - n * H;
  const float* x  = X + (size_t)n * H * C + (size_t)h * C;
  const float* av = a + (size_t)h * C;
  float s = 0.f;
  for (int c = lane; c < C; c += 32) s += x[c] * av[c];
  for (int off = 16; off; off >>= 1) s += __shfl_down(s, off, 32);
  if (lane == 0) out[warp] = s;
}

// ---------------- edge softmax (segment over dst) ----------------

__device__ __forceinline__ void edge_sd(const int* __restrict__ edge, int E,
                                        int i, int& s, int& d) {
  if (i < E) { s = edge[i]; d = edge[E + i]; }
  else       { s = d = i - E; }                 // appended self loops
}

__device__ __forceinline__ float leaky02(float x) {
  return x >= 0.f ? x : 0.2f * x;
}

__device__ __forceinline__ void atomicMaxF(float* addr, float v) {
  if (v >= 0.f) atomicMax((int*)addr, __float_as_int(v));
  else          atomicMin((unsigned*)addr, __float_as_uint(v));
}

__global__ void k_edge_max(const int* __restrict__ edge, int E, int Etot,
                           const float* __restrict__ es, const float* __restrict__ ed,
                           float* __restrict__ m, int H) {
  int i = blockIdx.x * blockDim.x + threadIdx.x;
  if (i >= Etot) return;
  int s, d; edge_sd(edge, E, i, s, d);
  for (int h = 0; h < H; ++h) {
    float e = leaky02(es[(size_t)s * H + h] + ed[(size_t)d * H + h]);
    atomicMaxF(&m[(size_t)d * H + h], e);
  }
}

__global__ void k_edge_expsum(const int* __restrict__ edge, int E, int Etot,
                              const float* __restrict__ es, const float* __restrict__ ed,
                              const float* __restrict__ m, float* __restrict__ den,
                              float* __restrict__ alpha, int H) {
  int i = blockIdx.x * blockDim.x + threadIdx.x;
  if (i >= Etot) return;
  int s, d; edge_sd(edge, E, i, s, d);
  for (int h = 0; h < H; ++h) {
    float e  = leaky02(es[(size_t)s * H + h] + ed[(size_t)d * H + h]);
    float ex = expf(e - m[(size_t)d * H + h]);
    alpha[(size_t)i * H + h] = ex;
    atomicAdd(&den[(size_t)d * H + h], ex);
  }
}

__global__ void k_edge_norm(const int* __restrict__ edge, int E, int Etot,
                            const float* __restrict__ den, float* __restrict__ alpha,
                            int H) {
  int i = blockIdx.x * blockDim.x + threadIdx.x;
  if (i >= Etot) return;
  int s, d; edge_sd(edge, E, i, s, d);
  for (int h = 0; h < H; ++h)
    alpha[(size_t)i * H + h] /= (den[(size_t)d * H + h] + 1e-16f);
}

// out[dst, j] += alpha[i, j/C] * X[src, j]   over all (edge, j) pairs
__global__ void k_edge_scatter(const int* __restrict__ edge, int E, int Etot,
                               const float* __restrict__ X, const float* __restrict__ alpha,
                               float* __restrict__ out, int H, int C) {
  long long t = (long long)blockIdx.x * blockDim.x + threadIdx.x;
  long long total = (long long)Etot * H * C;
  if (t >= total) return;
  int HC = H * C;
  int i = (int)(t / HC);
  int j = (int)(t - (long long)i * HC);
  int h = j / C;
  int s, d; edge_sd(edge, E, i, s, d);
  float al = alpha[(size_t)i * H + h];
  atomicAdd(&out[(size_t)d * HC + j], al * X[(size_t)s * HC + j]);
}

// ---------------- elementwise: x = elu(x + b) ----------------

__global__ void k_bias_elu(float* __restrict__ x, const float* __restrict__ b,
                           long long n, int HC) {
  long long t = (long long)blockIdx.x * blockDim.x + threadIdx.x;
  if (t >= n) return;
  int j = (int)(t % HC);
  float v = x[t] + b[j];
  x[t] = v > 0.f ? v : (expf(v) - 1.f);
}

// ---------------- final: residual + biases + LayerNorm(256) ----------------

__global__ void __launch_bounds__(256)
k_final_ln(const float* __restrict__ out2, const float* __restrict__ R,
           const float* __restrict__ b2, const float* __restrict__ bres,
           const float* __restrict__ gamma, const float* __restrict__ beta,
           float* __restrict__ y, int N) {
  int n = blockIdx.x;
  int c = threadIdx.x;                           // 256 channels, 8 waves
  float v = out2[(size_t)n * 256 + c] + b2[c] + R[(size_t)n * 256 + c] + bres[c];
  float s = v, q = v * v;
  for (int off = 16; off; off >>= 1) {
    s += __shfl_down(s, off, 32);
    q += __shfl_down(q, off, 32);
  }
  __shared__ float ss[8], qq[8];
  int w = threadIdx.x >> 5, l = threadIdx.x & 31;
  if (l == 0) { ss[w] = s; qq[w] = q; }
  __syncthreads();
  if (w == 0) {
    float s2 = (l < 8) ? ss[l] : 0.f;
    float q2 = (l < 8) ? qq[l] : 0.f;
    for (int off = 4; off; off >>= 1) {
      s2 += __shfl_down(s2, off, 32);
      q2 += __shfl_down(q2, off, 32);
    }
    if (l == 0) { ss[0] = s2; qq[0] = q2; }
  }
  __syncthreads();
  float mu  = ss[0] * (1.f / 256.f);
  float var = qq[0] * (1.f / 256.f) - mu * mu;
  y[(size_t)n * 256 + c] = (v - mu) * rsqrtf(var + 1e-5f) * gamma[c] + beta[c];
}

// ---------------------------------------------------------------------------

extern "C" void kernel_launch(void* const* d_in, const int* in_sizes, int n_in,
                              void* d_out, int out_size, void* d_ws, size_t ws_size,
                              hipStream_t stream) {
  const int*   edge     = (const int*)  d_in[0];   // [2,E] int32
  const float* emb      = (const float*)d_in[1];   // [N,128]
  const float* W1       = (const float*)d_in[2];   // [128,512]
  const float* att_src1 = (const float*)d_in[3];   // [4,128]
  const float* att_dst1 = (const float*)d_in[4];   // [4,128]
  const float* b1       = (const float*)d_in[5];   // [512]
  const float* W2       = (const float*)d_in[6];   // [512,256]
  const float* att_src2 = (const float*)d_in[7];   // [1,256]
  const float* att_dst2 = (const float*)d_in[8];   // [1,256]
  const float* b2       = (const float*)d_in[9];   // [256]
  const float* Wres     = (const float*)d_in[10];  // [128,256]
  const float* bres     = (const float*)d_in[11];  // [256]
  const float* gamma    = (const float*)d_in[12];  // [256]
  const float* beta     = (const float*)d_in[13];  // [256]
  float* out = (float*)d_out;

  const int EMB = 128, HID = 128, HEADS = 4, OUT = 256, HC1 = HEADS * HID; // 512
  const int E    = in_sizes[0] / 2;
  const int N    = in_sizes[1] / EMB;
  const int Etot = E + N;

  // ---- carve workspace (256B aligned regions) ----
  char* w = (char*)d_ws;
  auto carve = [&](size_t bytes) -> char* {
    char* p = w; w += (bytes + 255) & ~(size_t)255; return p;
  };
  float*  H1     = (float*) carve((size_t)N * HC1 * 4);   // reused: H2 | Rres
  float*  out1   = (float*) carve((size_t)N * HC1 * 4);   // -> x1 -> reused as out2
  __bf16* embb   = (__bf16*)carve((size_t)N * EMB * 2);
  __bf16* x1b    = (__bf16*)carve((size_t)N * HC1 * 2);
  __bf16* W1t    = (__bf16*)carve((size_t)HC1 * EMB * 2);
  __bf16* W2t    = (__bf16*)carve((size_t)OUT * HC1 * 2);
  __bf16* Wrt    = (__bf16*)carve((size_t)OUT * EMB * 2);
  float*  es1    = (float*) carve((size_t)N * HEADS * 4);
  float*  ed1    = (float*) carve((size_t)N * HEADS * 4);
  float*  m1     = (float*) carve((size_t)N * HEADS * 4);
  float*  den1   = (float*) carve((size_t)N * HEADS * 4);
  float*  alpha1 = (float*) carve((size_t)Etot * HEADS * 4);
  float*  es2    = (float*) carve((size_t)N * 4);
  float*  ed2    = (float*) carve((size_t)N * 4);
  float*  m2     = (float*) carve((size_t)N * 4);
  float*  den2   = (float*) carve((size_t)N * 4);
  float*  alpha2 = (float*) carve((size_t)Etot * 4);
  float*  H2     = H1;                          // [N,256] (H1 dead after layer1)
  float*  Rres   = H1 + (size_t)N * OUT;        // [N,256]
  float*  out2   = out1;                        // [N,256] (x1 dead after bf16 cvt)

  const unsigned NEG_INF = 0xFF800000u;
  auto blk = [](long long n, int b) { return (unsigned)((n + b - 1) / b); };

  // ---- stage bf16 operands ----
  k_f32_to_bf16<<<blk((long long)N * EMB, 256), 256, 0, stream>>>(emb, embb, (long long)N * EMB);
  k_transpose_bf16<<<blk(EMB * HC1, 256), 256, 0, stream>>>(W1,   W1t, EMB, HC1);
  k_transpose_bf16<<<blk(HC1 * OUT, 256), 256, 0, stream>>>(W2,   W2t, HC1, OUT);
  k_transpose_bf16<<<blk(EMB * OUT, 256), 256, 0, stream>>>(Wres, Wrt, EMB, OUT);

  // ---- layer 1: H1 = emb @ W1  (WMMA) ----
  {
    dim3 g(HC1 / 64, (N + 15) / 16);
    k_gemm_bf16_wmma<<<g, 128, 0, stream>>>(embb, W1t, H1, N, EMB, HC1);
  }
  k_dot_att<<<blk((long long)N * HEADS * 32, 256), 256, 0, stream>>>(H1, att_src1, es1, N, HEADS, HID);
  k_dot_att<<<blk((long long)N * HEADS * 32, 256), 256, 0, stream>>>(H1, att_dst1, ed1, N, HEADS, HID);

  k_fill_u32<<<blk((long long)N * HEADS, 256), 256, 0, stream>>>((unsigned*)m1,   NEG_INF, (long long)N * HEADS);
  k_fill_u32<<<blk((long long)N * HEADS, 256), 256, 0, stream>>>((unsigned*)den1, 0u,      (long long)N * HEADS);
  k_fill_u32<<<blk((long long)N * HC1,   256), 256, 0, stream>>>((unsigned*)out1, 0u,      (long long)N * HC1);

  k_edge_max   <<<blk(Etot, 256), 256, 0, stream>>>(edge, E, Etot, es1, ed1, m1, HEADS);
  k_edge_expsum<<<blk(Etot, 256), 256, 0, stream>>>(edge, E, Etot, es1, ed1, m1, den1, alpha1, HEADS);
  k_edge_norm  <<<blk(Etot, 256), 256, 0, stream>>>(edge, E, Etot, den1, alpha1, HEADS);
  k_edge_scatter<<<blk((long long)Etot * HC1, 256), 256, 0, stream>>>(
      edge, E, Etot, H1, alpha1, out1, HEADS, HID);

  // x1 = elu(out1 + b1)  (in place), then bf16 stage
  k_bias_elu<<<blk((long long)N * HC1, 256), 256, 0, stream>>>(out1, b1, (long long)N * HC1, HC1);
  k_f32_to_bf16<<<blk((long long)N * HC1, 256), 256, 0, stream>>>(out1, x1b, (long long)N * HC1);

  // ---- layer 2 GEMM + residual GEMM (WMMA) ----
  {
    dim3 g(OUT / 64, (N + 15) / 16);
    k_gemm_bf16_wmma<<<g, 128, 0, stream>>>(x1b,  W2t, H2,   N, HC1, OUT);
    k_gemm_bf16_wmma<<<g, 128, 0, stream>>>(embb, Wrt, Rres, N, EMB, OUT);
  }
  k_dot_att<<<blk((long long)N * 32, 256), 256, 0, stream>>>(H2, att_src2, es2, N, 1, OUT);
  k_dot_att<<<blk((long long)N * 32, 256), 256, 0, stream>>>(H2, att_dst2, ed2, N, 1, OUT);

  k_fill_u32<<<blk(N, 256), 256, 0, stream>>>((unsigned*)m2,   NEG_INF, N);
  k_fill_u32<<<blk(N, 256), 256, 0, stream>>>((unsigned*)den2, 0u,      N);
  k_fill_u32<<<blk((long long)N * OUT, 256), 256, 0, stream>>>((unsigned*)out2, 0u, (long long)N * OUT);

  k_edge_max   <<<blk(Etot, 256), 256, 0, stream>>>(edge, E, Etot, es2, ed2, m2, 1);
  k_edge_expsum<<<blk(Etot, 256), 256, 0, stream>>>(edge, E, Etot, es2, ed2, m2, den2, alpha2, 1);
  k_edge_norm  <<<blk(Etot, 256), 256, 0, stream>>>(edge, E, Etot, den2, alpha2, 1);
  k_edge_scatter<<<blk((long long)Etot * OUT, 256), 256, 0, stream>>>(
      edge, E, Etot, H2, alpha2, out2, 1, OUT);

  // ---- residual + LayerNorm -> d_out ----
  k_final_ln<<<N, 256, 0, stream>>>(out2, Rres, b2, bres, gamma, beta, out, N);
}
